// Attention_9285719294029
// MI455X (gfx1250) — matrix operand order repaired and checked by reading, not compiled, not run on hardware
//
#include <hip/hip_runtime.h>

typedef unsigned short u16;
typedef __attribute__((ext_vector_type(16))) __bf16 bf16x16;
typedef __attribute__((ext_vector_type(8)))  float  f32x8;
typedef __attribute__((ext_vector_type(4)))  int    v4i;

#define BB 2
#define SS 2048
#define DD 2048
#define HH 16
#define KVH 4
#define HD 128

#if defined(__HIP_DEVICE_COMPILE__) && __has_builtin(__builtin_amdgcn_global_load_async_to_lds_b128)
#define HAVE_ASYNC_LDS 1
#else
#define HAVE_ASYNC_LDS 0
#endif

__device__ __forceinline__ u16 f2bf(float f) {
    unsigned int u = __float_as_uint(f);
    unsigned int r = u + 0x7fffu + ((u >> 16) & 1u);
    return (u16)(r >> 16);
}
__device__ __forceinline__ float bf2f(u16 h) {
    return __uint_as_float(((unsigned int)h) << 16);
}

// ---- CDNA5 async global->LDS copy (ASYNCcnt-tracked, no VGPR round trip) ----
__device__ __forceinline__ void async_copy_b128(const u16* g, u16* l) {
#if HAVE_ASYNC_LDS
    // builtin signature: (v4i addrspace(1)*, v4i addrspace(3)*, imm offset, imm cpol)
    __builtin_amdgcn_global_load_async_to_lds_b128(
        (__attribute__((address_space(1))) v4i*)(unsigned long long)g,
        (__attribute__((address_space(3))) v4i*)(unsigned int)(unsigned long long)l,
        0, 0);
#else
    *reinterpret_cast<uint4*>(l) = *reinterpret_cast<const uint4*>(g);
#endif
}
__device__ __forceinline__ void async_wait() {
#if HAVE_ASYNC_LDS
#if __has_builtin(__builtin_amdgcn_s_wait_asynccnt)
    __builtin_amdgcn_s_wait_asynccnt(0);
#else
    asm volatile("s_wait_asynccnt 0x0" ::: "memory");
#endif
#endif
}

// build a 16-element bf16 fragment from two 16-byte chunks
__device__ __forceinline__ bf16x16 ld_frag(const u16* p0, const u16* p1) {
    bf16x16 r;
    reinterpret_cast<uint4*>(&r)[0] = *reinterpret_cast<const uint4*>(p0);
    reinterpret_cast<uint4*>(&r)[1] = *reinterpret_cast<const uint4*>(p1);
    return r;
}

__device__ __forceinline__ f32x8 fzero() {
    f32x8 z;
#pragma unroll
    for (int i = 0; i < 8; ++i) z[i] = 0.0f;
    return z;
}

// ---------------------------------------------------------------- convert
__global__ __launch_bounds__(256) void cvt_f32_bf16(const float* __restrict__ s,
                                                    u16* __restrict__ d, long n4) {
    long i = (long)blockIdx.x * blockDim.x + threadIdx.x;
    if (i >= n4) return;
    float4 f = reinterpret_cast<const float4*>(s)[i];
    ushort4 u;
    u.x = f2bf(f.x); u.y = f2bf(f.y); u.z = f2bf(f.z); u.w = f2bf(f.w);
    reinterpret_cast<ushort4*>(d)[i] = u;
}

// ---------------------------------------------------------------- RoPE (in place, bf16)
__global__ __launch_bounds__(256) void rope_kernel(u16* __restrict__ p, int nh, long total) {
    long i = (long)blockIdx.x * blockDim.x + threadIdx.x;
    if (i >= total) return;
    int  ii = (int)(i & 63);          // 0..63  (HD/2 pair index)
    long r  = i >> 6;
    int  h  = (int)(r % nh);
    long bs = r / nh;
    long s  = bs % SS;
    size_t base = (size_t)bs * ((size_t)nh * HD) + (size_t)h * HD;
    float inv = __powf(10000.0f, -(float)(2 * ii) / (float)HD);
    float ang = (float)s * inv;
    float sn, cs;
    __sincosf(ang, &sn, &cs);
    float x1 = bf2f(p[base + ii]);
    float x2 = bf2f(p[base + ii + 64]);
    p[base + ii]      = f2bf(x1 * cs - x2 * sn);
    p[base + ii + 64] = f2bf(x2 * cs + x1 * sn);
}

// ---------------------------------------------------------------- bf16 GEMM (C = A[M,K] * W[K,N])
// block tile 64x128, 8 waves of 32x32, K-step 32, WMMA f32_16x16x32_bf16
template <bool OUT_F32>
__global__ __launch_bounds__(256) void gemm_bf16(const u16* __restrict__ A,
                                                 const u16* __restrict__ W,
                                                 void* __restrict__ Cout,
                                                 int M, int N, int K) {
    __shared__ u16 As[64][40];    // pitch 40 halfwords = 80B (16B aligned rows)
    __shared__ u16 Bs[128][40];   // transposed: Bs[n][k]

    const int t    = threadIdx.x;
    const int lane = t & 31;
    const int wid  = t >> 5;
    const int ln16 = lane & 15;
    const int hi   = lane >> 4;               // lane half
    const int m0   = blockIdx.y * 64;
    const int n0   = blockIdx.x * 128;
    const int wm   = (wid >> 2) * 32;
    const int wn   = (wid & 3) * 32;

    f32x8 acc[2][2];
    acc[0][0] = fzero(); acc[0][1] = fzero();
    acc[1][0] = fzero(); acc[1][1] = fzero();

    for (int k0 = 0; k0 < K; k0 += 32) {
        // A tile 64x32 : one async b128 per thread (CDNA5 ASYNCcnt path)
        {
            int m  = t >> 2;
            int kq = (t & 3) * 8;
            async_copy_b128(&A[(size_t)(m0 + m) * K + k0 + kq], &As[m][kq]);
        }
        // W tile 32x128 -> transposed into Bs[n][k] (VGPR round trip: transpose)
#pragma unroll
        for (int i = 0; i < 8; ++i) {
            int pidx = t + i * 256;           // 0..2047 (pairs)
            int k = pidx >> 6;                // 0..31
            int n = (pidx & 63) * 2;          // 0..126
            unsigned int w =
                *reinterpret_cast<const unsigned int*>(&W[(size_t)(k0 + k) * N + n0 + n]);
            Bs[n][k]     = (u16)(w & 0xffffu);
            Bs[n + 1][k] = (u16)(w >> 16);
        }
        if (k0 + 32 < K) {  // speculative prefetch of next W tile -> global_prefetch_b8
            __builtin_prefetch(&W[(size_t)(k0 + 32 + (t >> 3)) * N + n0 + (t & 7) * 16], 0, 0);
        }
        async_wait();
        __syncthreads();

        const int kb0 = hi ? 8 : 0;           // A fragment K base (lane-half dependent)
        bf16x16 af[2], bw[2];
#pragma unroll
        for (int i = 0; i < 2; ++i) {
            int r = wm + i * 16 + ln16;
            af[i] = ld_frag(&As[r][kb0], &As[r][kb0 + 16]);
        }
        const int kb2 = hi * 8;               // B fragment K base
#pragma unroll
        for (int j = 0; j < 2; ++j) {
            int n = wn + j * 16 + ln16;
            bw[j] = ld_frag(&Bs[n][kb2], &Bs[n][kb2 + 16]);
        }
#pragma unroll
        for (int i = 0; i < 2; ++i)
#pragma unroll
            for (int j = 0; j < 2; ++j)
                acc[i][j] = __builtin_amdgcn_wmma_f32_16x16x32_bf16(
                    false, af[i], false, bw[j], (short)0, acc[i][j], false, false);
        __syncthreads();
    }

    // C layout: VGPR j -> M = j (lanes 0-15) / j+8 (lanes 16-31), N = lane%16
#pragma unroll
    for (int i = 0; i < 2; ++i)
#pragma unroll
        for (int j = 0; j < 2; ++j) {
            const int row0 = m0 + wm + i * 16 + hi * 8;
            const int col  = n0 + wn + j * 16 + ln16;
            if constexpr (OUT_F32) {
                float* p = (float*)Cout + (size_t)row0 * N + col;
#pragma unroll
                for (int vg = 0; vg < 8; ++vg) { *p = acc[i][j][vg]; p += N; }
            } else {
                u16* p = (u16*)Cout + (size_t)row0 * N + col;
#pragma unroll
                for (int vg = 0; vg < 8; ++vg) { *p = f2bf(acc[i][j][vg]); p += N; }
            }
        }
}

// ---------------------------------------------------------------- flash attention
// grid (S/64, H, B), 128 threads (4 waves). wave w owns q rows [q0+16w, q0+16w+16)
__global__ __launch_bounds__(128) void attn_kernel(const u16* __restrict__ qb,
                                                   const u16* __restrict__ kb,
                                                   const u16* __restrict__ vb,
                                                   u16* __restrict__ ob) {
    __shared__ u16 Ks[32][136];       // K tile, row-major [kpos][hd]
    __shared__ u16 Vt[128][40];       // V tile transposed [hd][kpos]
    __shared__ u16 Pw[4][16][40];     // per-wave probabilities, row-major [row][kpos]

    const int t    = threadIdx.x;
    const int lane = t & 31;
    const int wid  = t >> 5;
    const int ln16 = lane & 15;
    const int hi   = lane >> 4;
    const int b    = blockIdx.z;
    const int h    = blockIdx.y;
    const int q0   = blockIdx.x * 64;
    const int kvh  = h >> 2;                  // REP = 4
    const int qr0  = q0 + wid * 16;
    const float scale = 0.08838834764831845f; // 1/sqrt(128)

    // resident Q fragments (4 K-chunks of 32 over HD=128)
    bf16x16 qf[4];
    {
        int srow = qr0 + ln16;
        const u16* qrow = qb + (size_t)(b * SS + srow) * (HH * HD) + h * HD;
        int kb0 = hi ? 8 : 0;
#pragma unroll
        for (int c = 0; c < 4; ++c)
            qf[c] = ld_frag(qrow + c * 32 + kb0, qrow + c * 32 + kb0 + 16);
    }

    f32x8 o[8];
    float mrun[8], lrun[8];
#pragma unroll
    for (int n = 0; n < 8; ++n) o[n] = fzero();
#pragma unroll
    for (int j = 0; j < 8; ++j) { mrun[j] = -1e30f; lrun[j] = 0.0f; }

    const int nkt = q0 / 32 + 2;              // causal tile limit for the block

    for (int kt = 0; kt < nkt; ++kt) {
        // ---- K tile (32x128): async global->LDS b128, 4 chunks/thread
#pragma unroll
        for (int i = 0; i < 4; ++i) {
            int chunk = t + i * 128;          // 0..511 uint4 chunks
            int row = chunk >> 4;
            int c8  = (chunk & 15) * 8;
            const u16* src = kb + (size_t)(b * SS + kt * 32 + row) * (KVH * HD) + kvh * HD + c8;
            async_copy_b128(src, &Ks[row][c8]);
        }
        // ---- V tile transposed (VGPR round trip: transpose)
#pragma unroll
        for (int i = 0; i < 16; ++i) {
            int pidx = t + i * 128;           // 0..2047 pairs
            int kk = pidx >> 6;
            int d  = (pidx & 63) * 2;
            const u16* src = vb + (size_t)(b * SS + kt * 32 + kk) * (KVH * HD) + kvh * HD + d;
            unsigned int w = *reinterpret_cast<const unsigned int*>(src);
            Vt[d][kk]     = (u16)(w & 0xffffu);
            Vt[d + 1][kk] = (u16)(w >> 16);
        }
        async_wait();
        __syncthreads();

        // ---- scores: S = Q * K^T  (16x32 tile = 2 column halves x 4 K-chunks)
        f32x8 sc[2];
        sc[0] = fzero(); sc[1] = fzero();
#pragma unroll
        for (int hf = 0; hf < 2; ++hf) {
            int krow = hf * 16 + ln16;        // B operand: N = k position
#pragma unroll
            for (int c = 0; c < 4; ++c) {
                int db = c * 32 + hi * 8;
                bf16x16 bfr = ld_frag(&Ks[krow][db], &Ks[krow][db + 16]);
                sc[hf] = __builtin_amdgcn_wmma_f32_16x16x32_bf16(
                    false, qf[c], false, bfr, (short)0, sc[hf], false, false);
            }
        }

        // ---- online softmax (rows split by lane half; N = ln16)
        float alpha[8];
#pragma unroll
        for (int j = 0; j < 8; ++j) {
            int rl   = hi ? j + 8 : j;
            int grow = qr0 + rl;
            int c0 = kt * 32 + ln16;
            int c1 = c0 + 16;
            float s0 = sc[0][j] * scale + (c0 > grow ? -1e9f : 0.0f);
            float s1 = sc[1][j] * scale + (c1 > grow ? -1e9f : 0.0f);
            float tmax = fmaxf(s0, s1);
#pragma unroll
            for (int off = 1; off < 16; off <<= 1)
                tmax = fmaxf(tmax, __shfl_xor(tmax, off, 32));
            float mnew = fmaxf(mrun[j], tmax);
            alpha[j] = __expf(mrun[j] - mnew);
            float p0 = __expf(s0 - mnew);
            float p1 = __expf(s1 - mnew);
            float rs = p0 + p1;
#pragma unroll
            for (int off = 1; off < 16; off <<= 1)
                rs += __shfl_xor(rs, off, 32);
            lrun[j] = lrun[j] * alpha[j] + rs;
            mrun[j] = mnew;
            Pw[wid][rl][ln16]      = f2bf(p0);
            Pw[wid][rl][16 + ln16] = f2bf(p1);
        }
        __syncthreads();                      // P visible (C-layout -> A-layout relayout)

        // ---- O = diag(alpha)*O + P * V
        bf16x16 pf;
        {
            int kb0 = hi ? 8 : 0;
            pf = ld_frag(&Pw[wid][ln16][kb0], &Pw[wid][ln16][kb0 + 16]);
        }
        int kb2 = hi * 8;
#pragma unroll
        for (int n = 0; n < 8; ++n) {
#pragma unroll
            for (int j = 0; j < 8; ++j) o[n][j] *= alpha[j];
            int d = n * 16 + ln16;
            bf16x16 vf = ld_frag(&Vt[d][kb2], &Vt[d][kb2 + 16]);
            o[n] = __builtin_amdgcn_wmma_f32_16x16x32_bf16(
                false, pf, false, vf, (short)0, o[n], false, false);
        }
        __syncthreads();                      // before tiles are overwritten
    }

    // ---- epilogue: O /= l, store bf16 at [b, s, h*HD + d]
    {
        const int rlb  = hi * 8;
        u16* obase = ob + (size_t)(b * SS + qr0 + rlb) * (HH * HD) + h * HD + ln16;
#pragma unroll
        for (int j = 0; j < 8; ++j) {
            u16* p = obase + (size_t)j * (HH * HD);
            float inv_l = 1.0f / lrun[j];
#pragma unroll
            for (int n = 0; n < 8; ++n)
                p[n * 16] = f2bf(o[n][j] * inv_l);
        }
    }
}

// ---------------------------------------------------------------- launch
extern "C" void kernel_launch(void* const* d_in, const int* in_sizes, int n_in,
                              void* d_out, int out_size, void* d_ws, size_t ws_size,
                              hipStream_t stream) {
    const float* x  = (const float*)d_in[0];
    const float* wq = (const float*)d_in[1];
    const float* wk = (const float*)d_in[2];
    const float* wv = (const float*)d_in[3];
    const float* wo = (const float*)d_in[4];
    // d_in[5] = mask: causal, recomputed inline in attn_kernel

    u16* ws = (u16*)d_ws;
    const size_t nx  = (size_t)BB * SS * DD;      // 8388608
    const size_t nwq = (size_t)DD * HH * HD;      // 4194304
    const size_t nwk = (size_t)DD * KVH * HD;     // 1048576
    u16* xb  = ws;
    u16* wqb = xb  + nx;
    u16* wkb = wqb + nwq;
    u16* wvb = wkb + nwk;
    u16* wob = wvb + nwk;
    u16* qbu = wob + nwq;
    u16* kbu = qbu + nx;                          // q is B*S*H*HD == nx
    u16* vbu = kbu + (size_t)BB * SS * KVH * HD;
    u16* obu = vbu + (size_t)BB * SS * KVH * HD;

    // 1) f32 -> bf16
    cvt_f32_bf16<<<(unsigned)((nx  / 4 + 255) / 256), 256, 0, stream>>>(x,  xb,  nx  / 4);
    cvt_f32_bf16<<<(unsigned)((nwq / 4 + 255) / 256), 256, 0, stream>>>(wq, wqb, nwq / 4);
    cvt_f32_bf16<<<(unsigned)((nwk / 4 + 255) / 256), 256, 0, stream>>>(wk, wkb, nwk / 4);
    cvt_f32_bf16<<<(unsigned)((nwk / 4 + 255) / 256), 256, 0, stream>>>(wv, wvb, nwk / 4);
    cvt_f32_bf16<<<(unsigned)((nwq / 4 + 255) / 256), 256, 0, stream>>>(wo, wob, nwq / 4);

    const int M = BB * SS;                        // 4096
    // 2) projections (bf16 out)
    gemm_bf16<false><<<dim3((HH * HD) / 128, M / 64), 256, 0, stream>>>(xb, wqb, qbu, M, HH * HD, DD);
    gemm_bf16<false><<<dim3((KVH * HD) / 128, M / 64), 256, 0, stream>>>(xb, wkb, kbu, M, KVH * HD, DD);
    gemm_bf16<false><<<dim3((KVH * HD) / 128, M / 64), 256, 0, stream>>>(xb, wvb, vbu, M, KVH * HD, DD);

    // 3) RoPE on q and k
    {
        long tq = (long)BB * SS * HH * (HD / 2);
        long tk = (long)BB * SS * KVH * (HD / 2);
        rope_kernel<<<(unsigned)((tq + 255) / 256), 256, 0, stream>>>(qbu, HH, tq);
        rope_kernel<<<(unsigned)((tk + 255) / 256), 256, 0, stream>>>(kbu, KVH, tk);
    }

    // 4) flash attention
    attn_kernel<<<dim3(SS / 64, HH, BB), 128, 0, stream>>>(qbu, kbu, vbu, obu);

    // 5) output projection -> f32 d_out
    gemm_bf16<true><<<dim3(DD / 128, M / 64), 256, 0, stream>>>(obu, wob, (float*)d_out, M, DD, HH * HD);
}
